// MultiHeadAttention_9045201125399
// MI455X (gfx1250) — matrix operand order repaired and checked by reading, not compiled, not run on hardware
//
#include <hip/hip_runtime.h>
#include <hip/hip_bf16.h>
#include <math.h>

// ---------------------------------------------------------------------------
// MHA forward for MI455X (gfx1250, wave32, WMMA + async TDM).
//   D_MODEL=1024, HEADS=16, D_K=64, B=4, S=2048
// Pipeline (bf16 compute, f32 accumulate):
//   1) LDS-tiled transpose+convert Wq/Wk/Wv/Wo -> Wt bf16 [N,K]
//   2) vectorized convert query/key/value f32 -> bf16
//   3) QKV projection GEMMs via v_wmma_f32_16x16x32_bf16 (64x64 tile / wave)
//   4) LDS-tiled transpose V -> Vt [B,H,Dk,S] bf16
//   5) flash attention: block = (b,h,128 q rows); K/V chunks double-buffered
//      in LDS, prefetched by the Tensor Data Mover one chunk ahead of compute
//      (s_wait_tensorcnt + 1 barrier/chunk), WMMA scores, base-2 online
//      softmax (ds_swizzle xor-reductions), WMMA context
//   6) output projection GEMM (f32 out + bias)
// ---------------------------------------------------------------------------

#define D_MODEL 1024
#define HEADS   16
#define DK      64
#define BATCH   4
#define SEQ     2048
#define MTOT    (BATCH * SEQ)   // 8192 rows

typedef __attribute__((ext_vector_type(16))) __bf16 v16bf;
typedef __attribute__((ext_vector_type(8)))  __bf16 v8bf;
typedef __attribute__((ext_vector_type(8)))  float  v8f;
typedef __attribute__((ext_vector_type(4)))  unsigned int u32x4;
typedef __attribute__((ext_vector_type(4)))  int          i32x4;
typedef __attribute__((ext_vector_type(8)))  int          i32x8;

union ABfrag { v16bf v; v8bf h[2]; };

#if defined(__has_builtin)
#if __has_builtin(__builtin_amdgcn_tensor_load_to_lds) && __has_builtin(__builtin_amdgcn_s_wait_tensorcnt)
#define HAVE_TDM 1
#endif
#endif
#ifndef HAVE_TDM
#define HAVE_TDM 0
#endif

// ---- ds_swizzle xor-shuffle (group-of-32: offset = and 0x1f | xor<<10) -----
template <int XORMASK>
__device__ __forceinline__ float swz_xor(float x) {
    return __int_as_float(
        __builtin_amdgcn_ds_swizzle(__float_as_int(x), 0x1f | (XORMASK << 10)));
}
__device__ __forceinline__ void rowreduce_max8(float (&v)[8]) {
#pragma unroll
    for (int r = 0; r < 8; ++r) v[r] = fmaxf(v[r], swz_xor<1>(v[r]));
#pragma unroll
    for (int r = 0; r < 8; ++r) v[r] = fmaxf(v[r], swz_xor<2>(v[r]));
#pragma unroll
    for (int r = 0; r < 8; ++r) v[r] = fmaxf(v[r], swz_xor<4>(v[r]));
#pragma unroll
    for (int r = 0; r < 8; ++r) v[r] = fmaxf(v[r], swz_xor<8>(v[r]));
}
__device__ __forceinline__ void rowreduce_sum8(float (&v)[8]) {
#pragma unroll
    for (int r = 0; r < 8; ++r) v[r] += swz_xor<1>(v[r]);
#pragma unroll
    for (int r = 0; r < 8; ++r) v[r] += swz_xor<2>(v[r]);
#pragma unroll
    for (int r = 0; r < 8; ++r) v[r] += swz_xor<4>(v[r]);
#pragma unroll
    for (int r = 0; r < 8; ++r) v[r] += swz_xor<8>(v[r]);
}

// ---- Tensor Data Mover: 2D tile (2-byte elements) global -> LDS ------------
__device__ __forceinline__ void tdm_load_2d_b16(unsigned lds_byte_addr,
                                                const void* gaddr,
                                                unsigned t_d0, unsigned t_d1,
                                                unsigned tile_d0, unsigned tile_d1,
                                                unsigned long long s0_elems) {
#if HAVE_TDM
    unsigned long long ga = (unsigned long long)(uintptr_t)gaddr;
    u32x4 g0;
    g0[0] = 1u;                                               // count=1 (valid), user mode
    g0[1] = lds_byte_addr;                                    // D#.lds_addr
    g0[2] = (unsigned)(ga & 0xffffffffu);                     // global_addr[31:0]
    g0[3] = (unsigned)((ga >> 32) & 0x1ffffffu) | (2u << 30); // addr[56:32] | type=2
    i32x8 g1;
    g1[0] = (int)(1u << 16);                                  // data_size=1 -> 2B elems
    g1[1] = (int)((t_d0 & 0xffffu) << 16);                    // tensor_dim0[15:0]
    g1[2] = (int)(((t_d0 >> 16) & 0xffffu) | ((t_d1 & 0xffffu) << 16));
    g1[3] = (int)(((t_d1 >> 16) & 0xffffu) | ((tile_d0 & 0xffffu) << 16));
    g1[4] = (int)(tile_d1 & 0xffffu);                         // tile_dim1 (tile_dim2=0)
    g1[5] = (int)(unsigned)(s0_elems & 0xffffffffull);        // dim0 stride[31:0]
    g1[6] = (int)(unsigned)((s0_elems >> 32) & 0xffffull);    // stride[47:32]
    g1[7] = 0;
    i32x4 gz = (i32x4)0;
#if defined(__clang_major__) && (__clang_major__ >= 23)
    i32x8 gz8 = (i32x8)0;
    __builtin_amdgcn_tensor_load_to_lds(g0, g1, gz, gz, gz8, 0);
#else
    __builtin_amdgcn_tensor_load_to_lds(g0, g1, gz, gz, 0);
#endif
#else
    (void)lds_byte_addr; (void)gaddr; (void)t_d0; (void)t_d1;
    (void)tile_d0; (void)tile_d1; (void)s0_elems;
#endif
}

// ---------------------------------------------------------------- converts --

__global__ __launch_bounds__(256) void k_f32_to_bf16(const float* __restrict__ x,
                                                     __bf16* __restrict__ y, int n) {
    int i = (blockIdx.x * blockDim.x + threadIdx.x) * 8;
    if (i + 8 <= n) {
        float4 a = *(const float4*)(x + i);
        float4 b = *(const float4*)(x + i + 4);
        v8bf o;
        o[0] = (__bf16)a.x; o[1] = (__bf16)a.y; o[2] = (__bf16)a.z; o[3] = (__bf16)a.w;
        o[4] = (__bf16)b.x; o[5] = (__bf16)b.y; o[6] = (__bf16)b.z; o[7] = (__bf16)b.w;
        *(v8bf*)(y + i) = o;
    }
}

// Wt[n*1024 + k] = (bf16)W[k*1024 + n] via 32x32 LDS tile (coalesced r/w)
__global__ __launch_bounds__(256) void k_transpose_w_bf16(const float* __restrict__ W,
                                                          __bf16* __restrict__ Wt) {
    __shared__ __bf16 tile[32][33];
    const int tx = threadIdx.x, ty = threadIdx.y;           // (32, 8)
    const int k0 = (blockIdx.x & 31) * 32;
    const int n0 = (blockIdx.x >> 5) * 32;
#pragma unroll
    for (int i = 0; i < 4; ++i)
        tile[ty * 4 + i][tx] = (__bf16)W[(size_t)(k0 + ty * 4 + i) * D_MODEL + n0 + tx];
    __syncthreads();
#pragma unroll
    for (int i = 0; i < 4; ++i)
        Wt[(size_t)(n0 + ty * 4 + i) * D_MODEL + k0 + tx] = tile[tx][ty * 4 + i];
}

// Vt[((b*H+h)*DK+d)*S + s] = Vp[(b*S+s)*D + h*DK + d] via 32x32 LDS tile
__global__ __launch_bounds__(256) void k_transpose_v_bf16(const __bf16* __restrict__ Vp,
                                                          __bf16* __restrict__ Vt) {
    __shared__ __bf16 tile[32][33];
    int t = blockIdx.x;                                     // B*H * (S/32) * (DK/32) = 8192
    const int d0 = (t & 1) * 32;  t >>= 1;
    const int s0 = (t & 63) * 32; t >>= 6;
    const int h = t & (HEADS - 1);
    const int b = t >> 4;
    const int tx = threadIdx.x, ty = threadIdx.y;           // (32, 8)
#pragma unroll
    for (int i = 0; i < 4; ++i)
        tile[ty * 4 + i][tx] =
            Vp[(size_t)(b * SEQ + s0 + ty * 4 + i) * D_MODEL + h * DK + d0 + tx];
    __syncthreads();
#pragma unroll
    for (int i = 0; i < 4; ++i)
        Vt[(size_t)((b * HEADS + h) * DK + d0 + ty * 4 + i) * SEQ + s0 + tx] =
            tile[tx][ty * 4 + i];
}

// ------------------------------------------------------------------- GEMM ---
// C[M,N] = A[M,K] @ Wt^T + bias ; A bf16 row-major, Wt bf16 [N,K].
// One wave computes a 64x64 tile with 4x4 f32 WMMA accumulators.
template <bool OUT_BF16>
__global__ __launch_bounds__(256)
__attribute__((amdgpu_waves_per_eu(2)))
void k_gemm_wmma_bf16(const __bf16* __restrict__ A, const __bf16* __restrict__ Wt,
                      const float* __restrict__ bias, void* __restrict__ Cout,
                      int M, int N, int K) {
    const int lane = threadIdx.x & 31;
    const int half = lane >> 4;
    const int l16  = lane & 15;
    const int wid  = blockIdx.x * (blockDim.x >> 5) + (threadIdx.x >> 5);
    const int ntiles64 = N >> 6;
    const int m0 = (wid / ntiles64) * 64;
    const int n0 = (wid % ntiles64) * 64;

    v8f acc[4][4];
#pragma unroll
    for (int i = 0; i < 4; ++i)
#pragma unroll
        for (int j = 0; j < 4; ++j) acc[i][j] = (v8f)0.0f;

    for (int kb = 0; kb < K; kb += 32) {
        ABfrag af[4];
        const __bf16* abase = A + (size_t)(m0 + l16) * K + kb + half * 8;
        if (kb + 32 < K)
            __builtin_prefetch(abase + 32, 0, 0);   // global_prefetch next k-step
#pragma unroll
        for (int mt = 0; mt < 4; ++mt) {
            const __bf16* p = abase + (size_t)mt * 16 * K;
            af[mt].h[0] = *(const v8bf*)p;
            af[mt].h[1] = *(const v8bf*)(p + 16);
        }
#pragma unroll
        for (int nt = 0; nt < 4; ++nt) {
            const __bf16* bp = Wt + (size_t)(n0 + nt * 16 + l16) * K + kb + half * 8;
            ABfrag bf;
            bf.h[0] = *(const v8bf*)bp;
            bf.h[1] = *(const v8bf*)(bp + 16);
#pragma unroll
            for (int mt = 0; mt < 4; ++mt)
                acc[mt][nt] = __builtin_amdgcn_wmma_f32_16x16x32_bf16(
                    false, af[mt].v, false, bf.v, (short)0, acc[mt][nt], false, false);
        }
    }

#pragma unroll
    for (int nt = 0; nt < 4; ++nt) {
        const int col = n0 + nt * 16 + l16;
        const float bv = bias[col];
#pragma unroll
        for (int mt = 0; mt < 4; ++mt) {
#pragma unroll
            for (int r = 0; r < 8; ++r) {
                const int row = m0 + mt * 16 + half * 8 + r;
                const float val = acc[mt][nt][r] + bv;
                if (OUT_BF16)
                    ((__bf16*)Cout)[(size_t)row * N + col] = (__bf16)val;
                else
                    ((float*)Cout)[(size_t)row * N + col] = val;
            }
        }
    }
}

// -------------------------------------------------------- flash attention ---
// Block = 8 waves = one (b, h, 128-row q group). K/V chunks (32 keys) are
// double-buffered in LDS; the TDM fetches chunk i+1 while waves compute on
// chunk i (one workgroup barrier per chunk). Softmax is done in base 2.
__global__ __launch_bounds__(256)
__attribute__((amdgpu_waves_per_eu(1)))
void k_flash_attn(const __bf16* __restrict__ Qp, const __bf16* __restrict__ Kp,
                  const __bf16* __restrict__ Vt, const int* __restrict__ mask,
                  __bf16* __restrict__ Cx) {
    __shared__ __bf16 ldsK[2][32 * 64];   // K chunk: [32 keys][64 dk]  (2 x 4 KB)
    __shared__ __bf16 ldsV[2][64 * 32];   // V chunk: [64 d][32 keys]   (2 x 4 KB)
    __shared__ __bf16 plds[8][16 * 32];   // per-wave P tile (A layout) (8 KB)

    const int lane = threadIdx.x & 31;
    const int w    = threadIdx.x >> 5;
    const int half = lane >> 4;
    const int l16  = lane & 15;

    const int qg = blockIdx.x & 15;       // 16 q-groups of 128 rows
    const int bh = blockIdx.x >> 4;
    const int h  = bh & (HEADS - 1);
    const int b  = bh >> 4;
    const int q0 = qg * 128 + w * 16;

    const __bf16* kbase = Kp + (size_t)(b * SEQ) * D_MODEL + h * DK;
    const __bf16* vbase = Vt + (size_t)((b * HEADS + h) * DK) * SEQ;

    // Q fragments (A operand, Dk=64 -> 2 frags), loaded once
    ABfrag qf[2];
    {
        const __bf16* qbase = Qp + (size_t)(b * SEQ + q0 + l16) * D_MODEL + h * DK;
#pragma unroll
        for (int kk = 0; kk < 2; ++kk) {
            const __bf16* p = qbase + kk * 32 + half * 8;
            qf[kk].h[0] = *(const v8bf*)p;
            qf[kk].h[1] = *(const v8bf*)(p + 16);
        }
    }

    v8f ctx[4];
#pragma unroll
    for (int t = 0; t < 4; ++t) ctx[t] = (v8f)0.0f;
    float mrow[8], lrow[8];
#pragma unroll
    for (int r = 0; r < 8; ++r) { mrow[r] = -3.0e38f; lrow[r] = 0.0f; }

    const float SC_LOG2E = 0.125f * 1.44269504088896340736f;  // (1/sqrt(Dk))*log2(e)

#if HAVE_TDM
    if (w == 0) {   // prologue: DMA chunk 0 into buffer 0
        tdm_load_2d_b16((unsigned)(uintptr_t)&ldsK[0][0], kbase,
                        DK, SEQ, DK, 32, D_MODEL);
        tdm_load_2d_b16((unsigned)(uintptr_t)&ldsV[0][0], vbase,
                        SEQ, DK, 32, DK, SEQ);
    }
#endif

    for (int j0 = 0; j0 < SEQ; j0 += 32) {
        const int buf = (j0 >> 5) & 1;
        const __bf16* curK = ldsK[buf];
        const __bf16* curV = ldsV[buf];

#if HAVE_TDM
        if (w == 0) __builtin_amdgcn_s_wait_tensorcnt(0);  // chunk j0 landed
        __syncthreads();   // data visible to all waves; prev chunk fully consumed
        if (w == 0 && j0 + 32 < SEQ) {                     // DMA next chunk, overlapped
            const int nb = buf ^ 1;
            tdm_load_2d_b16((unsigned)(uintptr_t)&ldsK[nb][0],
                            kbase + (size_t)(j0 + 32) * D_MODEL,
                            DK, SEQ - (j0 + 32), DK, 32, D_MODEL);
            tdm_load_2d_b16((unsigned)(uintptr_t)&ldsV[nb][0],
                            vbase + (j0 + 32),
                            SEQ - (j0 + 32), DK, 32, DK, SEQ);
        }
#else
        {
            const int e = threadIdx.x * 8;                 // 256 thr x 16B = 4 KB each
            *(v8bf*)&ldsK[buf][e] =
                *(const v8bf*)(kbase + (size_t)(j0 + (e >> 6)) * D_MODEL + (e & 63));
            *(v8bf*)&ldsV[buf][e] =
                *(const v8bf*)(vbase + (size_t)(e >> 5) * SEQ + j0 + (e & 31));
        }
        __syncthreads();
#endif

        // ---- scores: two 16x16 tiles from LDS K (pre-scaled to base-2 domain)
        v8f sc[2];
#pragma unroll
        for (int t = 0; t < 2; ++t) {
            v8f s = (v8f)0.0f;
#pragma unroll
            for (int kb = 0; kb < 2; ++kb) {
                const __bf16* kp = &curK[(t * 16 + l16) * 64 + kb * 32 + half * 8];
                ABfrag kf;
                kf.h[0] = *(const v8bf*)kp;
                kf.h[1] = *(const v8bf*)(kp + 16);
                s = __builtin_amdgcn_wmma_f32_16x16x32_bf16(
                    false, qf[kb].v, false, kf.v, (short)0, s, false, false);
            }
            const int mv = mask[(size_t)b * SEQ + j0 + t * 16 + l16];
#pragma unroll
            for (int r = 0; r < 8; ++r)
                s[r] = (mv != 0) ? s[r] * SC_LOG2E : -1.0e9f;
            sc[t] = s;
        }

        // ---- online softmax in base 2 (rows live in 16-lane halves)
        float tmax[8];
#pragma unroll
        for (int r = 0; r < 8; ++r) tmax[r] = fmaxf(sc[0][r], sc[1][r]);
        rowreduce_max8(tmax);

        float p0[8], p1[8], alpha[8], rsum[8];
#pragma unroll
        for (int r = 0; r < 8; ++r) {
            const float mn = fmaxf(mrow[r], tmax[r]);
            alpha[r] = exp2f(mrow[r] - mn);
            mrow[r]  = mn;
            p0[r] = exp2f(sc[0][r] - mn);
            p1[r] = exp2f(sc[1][r] - mn);
            rsum[r] = p0[r] + p1[r];
        }
        rowreduce_sum8(rsum);
#pragma unroll
        for (int r = 0; r < 8; ++r) lrow[r] = lrow[r] * alpha[r] + rsum[r];
#pragma unroll
        for (int t = 0; t < 4; ++t)
#pragma unroll
            for (int r = 0; r < 8; ++r) ctx[t][r] *= alpha[r];

        // ---- P (C layout) -> per-wave LDS -> A-layout fragment
        __bf16* pl = plds[w];
#pragma unroll
        for (int r = 0; r < 8; ++r) {
            const int row = r + half * 8;
            pl[row * 32 + l16]      = (__bf16)p0[r];
            pl[row * 32 + 16 + l16] = (__bf16)p1[r];
        }
        asm volatile("s_wait_dscnt 0" ::: "memory");  // DS in-order per wave
        ABfrag pf;
        {
            const __bf16* pp = pl + l16 * 32 + half * 8;
            pf.h[0] = *(const v8bf*)pp;
            pf.h[1] = *(const v8bf*)(pp + 16);
        }

        // ---- context: ctx[16q x 64d] += P[16x32] @ V[32x64] from LDS V
#pragma unroll
        for (int t = 0; t < 4; ++t) {
            const __bf16* vp = &curV[(t * 16 + l16) * 32 + half * 8];
            ABfrag vf;
            vf.h[0] = *(const v8bf*)vp;
            vf.h[1] = *(const v8bf*)(vp + 16);
            ctx[t] = __builtin_amdgcn_wmma_f32_16x16x32_bf16(
                false, pf.v, false, vf.v, (short)0, ctx[t], false, false);
        }
#if !HAVE_TDM
        __syncthreads();   // protect single-rotation LDS before next overwrite
#endif
    }

    // ---- finalize: ctx / l, write bf16 context [B,S,D]
    float inv[8];
#pragma unroll
    for (int r = 0; r < 8; ++r) inv[r] = __builtin_amdgcn_rcpf(lrow[r]);
#pragma unroll
    for (int t = 0; t < 4; ++t)
#pragma unroll
        for (int r = 0; r < 8; ++r) {
            const int row = b * SEQ + q0 + half * 8 + r;
            const int col = h * DK + t * 16 + l16;
            Cx[(size_t)row * D_MODEL + col] = (__bf16)(ctx[t][r] * inv[r]);
        }
}

// ------------------------------------------------------------------ launch --

extern "C" void kernel_launch(void* const* d_in, const int* in_sizes, int n_in,
                              void* d_out, int out_size, void* d_ws, size_t ws_size,
                              hipStream_t stream) {
    (void)in_sizes; (void)n_in; (void)out_size; (void)ws_size;
    const float* query = (const float*)d_in[0];
    const float* key   = (const float*)d_in[1];
    const float* value = (const float*)d_in[2];
    const int*   mask  = (const int*)d_in[3];
    const float* Wq = (const float*)d_in[4];  const float* bq = (const float*)d_in[5];
    const float* Wk = (const float*)d_in[6];  const float* bk = (const float*)d_in[7];
    const float* Wv = (const float*)d_in[8];  const float* bv = (const float*)d_in[9];
    const float* Wo = (const float*)d_in[10]; const float* bo = (const float*)d_in[11];
    float* out = (float*)d_out;

    char* ws = (char*)d_ws;
    const size_t MB = 1u << 20;
    __bf16* WtQ = (__bf16*)(ws + 0 * MB);       // 2 MiB each
    __bf16* WtK = (__bf16*)(ws + 2 * MB);
    __bf16* WtV = (__bf16*)(ws + 4 * MB);
    __bf16* WtO = (__bf16*)(ws + 6 * MB);
    __bf16* Xq  = (__bf16*)(ws + 8 * MB);       // 16 MiB each
    __bf16* Xk  = (__bf16*)(ws + 24 * MB);
    __bf16* Xv  = (__bf16*)(ws + 40 * MB);
    __bf16* Qp  = (__bf16*)(ws + 56 * MB);
    __bf16* Kp  = (__bf16*)(ws + 72 * MB);
    __bf16* Vp  = (__bf16*)(ws + 88 * MB);
    __bf16* Vt  = (__bf16*)(ws + 104 * MB);
    __bf16* Cx  = (__bf16*)(ws + 120 * MB);

    const int nX = MTOT * D_MODEL;              // 8388608

    dim3 tb(32, 8);
    // 1) weight transpose+convert: 32x32 tiles over 1024x1024 -> 1024 blocks
    k_transpose_w_bf16<<<1024, tb, 0, stream>>>(Wq, WtQ);
    k_transpose_w_bf16<<<1024, tb, 0, stream>>>(Wk, WtK);
    k_transpose_w_bf16<<<1024, tb, 0, stream>>>(Wv, WtV);
    k_transpose_w_bf16<<<1024, tb, 0, stream>>>(Wo, WtO);
    // 2) activation convert (8 elems/thread, b128 in/out)
    k_f32_to_bf16<<<nX / (256 * 8), 256, 0, stream>>>(query, Xq, nX);
    k_f32_to_bf16<<<nX / (256 * 8), 256, 0, stream>>>(key,   Xk, nX);
    k_f32_to_bf16<<<nX / (256 * 8), 256, 0, stream>>>(value, Xv, nX);
    // 3) QKV projections: 2048 waves -> 256 blocks of 8 waves
    const int gemmBlocks = (MTOT / 64) * (D_MODEL / 64) / 8;
    k_gemm_wmma_bf16<true><<<gemmBlocks, 256, 0, stream>>>(Xq, WtQ, bq, Qp, MTOT, D_MODEL, D_MODEL);
    k_gemm_wmma_bf16<true><<<gemmBlocks, 256, 0, stream>>>(Xk, WtK, bk, Kp, MTOT, D_MODEL, D_MODEL);
    k_gemm_wmma_bf16<true><<<gemmBlocks, 256, 0, stream>>>(Xv, WtV, bv, Vp, MTOT, D_MODEL, D_MODEL);
    // 4) V transpose: B*H*(S/32)*(DK/32) = 8192 tiles
    k_transpose_v_bf16<<<8192, tb, 0, stream>>>(Vp, Vt);
    // 5) flash attention: B*H*16 q-groups = 1024 blocks of 8 waves
    k_flash_attn<<<BATCH * HEADS * 16, 256, 0, stream>>>(Qp, Kp, Vt, mask, Cx);
    // 6) output projection, f32 out + bias
    k_gemm_wmma_bf16<false><<<gemmBlocks, 256, 0, stream>>>(Cx, WtO, bo, out, MTOT, D_MODEL, D_MODEL);
}
